// TransformerClass_40020505264269
// MI455X (gfx1250) — matrix operand order repaired and checked by reading, not compiled, not run on hardware
//
#include <hip/hip_runtime.h>
#include <hip/hip_bf16.h>
#include <cstdint>
#include <cstddef>

// ---------------------------------------------------------------------------
// Types for CDNA5 WMMA (wave32): v_wmma_f32_16x16x32_bf16
// ---------------------------------------------------------------------------
typedef __attribute__((ext_vector_type(16))) __bf16 bhalf16;
typedef __attribute__((ext_vector_type(8)))  __bf16 bhalf8;
typedef __attribute__((ext_vector_type(8)))  float  floatx8;
typedef __attribute__((ext_vector_type(4)))  unsigned int uint4v;
typedef __attribute__((ext_vector_type(8)))  int int8v;
typedef __attribute__((ext_vector_type(4)))  int int4v;

#define WMMA_BF16(A, B, C) \
    __builtin_amdgcn_wmma_f32_16x16x32_bf16(false, (A), false, (B), (short)0, (C), false, false)

// A-matrix fragment (16x32 bf16): lane m = lane&15; lanes<16 hold K[0..15],
// lanes>=16 hold K[16..31].  p must already include the (lane>=16 ? +16 : 0).
__device__ __forceinline__ bhalf16 ldA16(const __bf16* p) {
    bhalf8 lo = *(const bhalf8*)p;
    bhalf8 hi = *(const bhalf8*)(p + 8);
    bhalf16 r;
#pragma unroll
    for (int i = 0; i < 8; ++i) { r[i] = lo[i]; r[i + 8] = hi[i]; }
    return r;
}

// B-matrix fragment (32x16 bf16): lane n = lane&15; lanes<16 hold K=0..15,
// lanes>=16 hold K=16..31 (16 contiguous elements). p must include (lane>=16 ? +16 : 0).
__device__ __forceinline__ bhalf16 ldB16(const __bf16* p) {
    bhalf8 lo = *(const bhalf8*)p;
    bhalf8 hi = *(const bhalf8*)(p + 8);
    bhalf16 r;
#pragma unroll
    for (int i = 0; i < 8; ++i) { r[i] = lo[i]; r[i + 8] = hi[i]; }
    return r;
}

// ---------------------------------------------------------------------------
// Tensor Data Mover: load a 2-D bf16 tile [rows=64, cols=KSTEP=32] from a
// row-major matrix (row stride = K elements) into LDS (packed 64x32).
// 6-arg builtin (clang-23 / therock-10.0):
//   (uint32x4 g0, int32x8 g1, int32x4 g2, int32x4 g3, int32x8 g4, i32 cpol).
// D# packing per CDNA5 ISA 8.3/8.4 (count=1, type=2 "image", data_size=2B).
// ---------------------------------------------------------------------------
__device__ __forceinline__ void tdm_load_tile(const __bf16* src, unsigned lds_off,
                                              int K, int N) {
    const unsigned long long gaddr = (unsigned long long)(uintptr_t)src;
    uint4v g0;
    g0[0] = 1u;                                     // count=1, user mode
    g0[1] = lds_off;                                // LDS byte address
    g0[2] = (unsigned)gaddr;                        // global_addr[31:0]
    g0[3] = (unsigned)(gaddr >> 32) | (2u << 30);   // global_addr[56:32] | type=2
    int8v g1;
    g1[0] = 1 << 16;                                // data_size=1 (2 bytes)
    g1[1] = (K & 0xffff) << 16;                     // tensor_dim0[15:0]
    g1[2] = ((K >> 16) & 0xffff) | ((N & 0xffff) << 16);   // dim0[31:16] | dim1[15:0]
    g1[3] = ((N >> 16) & 0xffff) | (32 << 16);      // dim1[31:16] | tile_dim0=32
    g1[4] = 64;                                     // tile_dim1=64, tile_dim2=0
    g1[5] = K;                                      // tensor_dim0_stride[31:0]
    g1[6] = 0;                                      // stride0[47:32] | stride1[15:0]
    g1[7] = 0;
    int4v z4 = {0, 0, 0, 0};
    int8v z8 = {0, 0, 0, 0, 0, 0, 0, 0};
    __builtin_amdgcn_tensor_load_to_lds(g0, g1, z4, z4, z8, 0);
}

// ---------------------------------------------------------------------------
// Weight transpose + bf16 convert:  W[K,N] f32 -> Wt[N,K] bf16
// ---------------------------------------------------------------------------
__global__ void wtr_kernel(const float* __restrict__ W, __bf16* __restrict__ Wt,
                           int K, int N) {
    int idx = blockIdx.x * blockDim.x + threadIdx.x;
    if (idx >= K * N) return;
    int k = idx / N, n = idx % N;
    Wt[(size_t)n * K + k] = (__bf16)W[idx];
}

// ---------------------------------------------------------------------------
// Embedding: h0[b,s,:] = emb[x[b,s]] + pos[s];  also bf16 copy.
// ---------------------------------------------------------------------------
__global__ void embed_kernel(const int* __restrict__ x, const float* __restrict__ emb,
                             const float* __restrict__ pos, float* __restrict__ h0,
                             __bf16* __restrict__ hb, int S, int D) {
    const int row = blockIdx.x;          // b*S + s
    const int s = row % S;
    const int tok = x[row];
    const float* e = emb + (size_t)tok * D;
    const float* p = pos + (size_t)s * D;
    const size_t base = (size_t)row * D;
    for (int d = threadIdx.x; d < D; d += blockDim.x) {
        float v = e[d] + p[d];
        h0[base + d] = v;
        hb[base + d] = (__bf16)v;
    }
}

// ---------------------------------------------------------------------------
// WMMA GEMM:  out[M,N] = A[M,K](bf16) @ Wt[N,K](bf16)^T + bias (+ resid)
// Block tile 256(M) x 64(N); 8 waves, each owning a 32x64 sub-tile
// (2 A-fragments x 4 B-fragments -> 8 WMMAs per k-step).
// The 64x32 bf16 weight panel is TDM-loaded into LDS (double buffered,
// s_wait_tensorcnt + barrier) and shared by all 8 waves.
// Optional outputs:
//   outF : f32 [M,N]     outB : bf16 [M,N]
//   outT : bf16 transposed per-batch: outT[(b*N + n)*S + s]   (for V)
// ---------------------------------------------------------------------------
__global__ void gemm_bf16(const __bf16* __restrict__ A, const __bf16* __restrict__ Wt,
                          const float* __restrict__ bias, const float* __restrict__ resid,
                          float* __restrict__ outF, __bf16* __restrict__ outB,
                          __bf16* __restrict__ outT,
                          int M, int N, int K, int S) {
    constexpr int KSTEP = 32;
    __shared__ __align__(16) __bf16 tileB[2][64 * KSTEP];   // 4 KB per buffer

    const int lane = threadIdx.x & 31;
    const int wv = threadIdx.x >> 5;
    const int tilesN = N / 64;
    const int bm = blockIdx.x / tilesN;
    const int bn = blockIdx.x % tilesN;
    const int row0 = bm * 256 + wv * 32;
    const int col0 = bn * 64;

    const int nloc = lane & 15;
    const int half8 = (lane >= 16) ? 8 : 0;
    const int half16 = (lane >= 16) ? 16 : 0;

    const __bf16* arow0 = A + (size_t)(row0 + nloc) * K + half16;
    const __bf16* arow1 = arow0 + (size_t)16 * K;
    const __bf16* wpanel = Wt + (size_t)col0 * K;   // 64 rows of Wt, stride K

    const unsigned lds0 = (unsigned)(uintptr_t)(&tileB[0][0]);
    const unsigned lds1 = (unsigned)(uintptr_t)(&tileB[1][0]);

    floatx8 acc[2][4];
#pragma unroll
    for (int s = 0; s < 2; ++s)
#pragma unroll
        for (int j = 0; j < 4; ++j) acc[s][j] = (floatx8){};

    if (wv == 0) tdm_load_tile(wpanel, lds0, K, N);

    int cur = 0;
    for (int k0 = 0; k0 < K; k0 += KSTEP) {
        __syncthreads();                 // everyone done reading buf[cur] (prev use)
        if (wv == 0) __builtin_amdgcn_s_wait_tensorcnt(0);
        __syncthreads();                 // tile ready for all waves
        if (wv == 0 && (k0 + KSTEP) < K)
            tdm_load_tile(wpanel + k0 + KSTEP, cur ? lds0 : lds1, K, N);

        bhalf16 a0 = ldA16(arow0 + k0);
        bhalf16 a1 = ldA16(arow1 + k0);
#pragma unroll
        for (int j = 0; j < 4; ++j) {
            bhalf16 bf = ldB16(&tileB[cur][(j * 16 + nloc) * KSTEP + half16]);
            acc[0][j] = WMMA_BF16(a0, bf, acc[0][j]);
            acc[1][j] = WMMA_BF16(a1, bf, acc[1][j]);
        }
        cur ^= 1;
    }

    // Epilogue: 8 independent 16x16 C tiles
#pragma unroll
    for (int sub = 0; sub < 2; ++sub) {
#pragma unroll
        for (int j = 0; j < 4; ++j) {
            const int nglob = col0 + j * 16 + nloc;
            const float bv = bias ? bias[nglob] : 0.0f;
            const int mbase = row0 + sub * 16 + half8;
            floatx8 c = acc[sub][j];
#pragma unroll
            for (int r = 0; r < 8; ++r) {
                float v = c[r] + bv;
                const size_t idx = (size_t)(mbase + r) * N + nglob;
                if (resid) v += resid[idx];
                if (outF) outF[idx] = v;
                if (outB) outB[idx] = (__bf16)v;
                c[r] = v;
            }
            if (outT) {
                const int bb = mbase / S;
                const int s0 = mbase % S;   // 8 consecutive tokens per lane
                bhalf8 st;
#pragma unroll
                for (int r = 0; r < 8; ++r) st[r] = (__bf16)c[r];
                *(bhalf8*)(outT + ((size_t)bb * N + nglob) * S + s0) = st;
            }
        }
    }
}

// ---------------------------------------------------------------------------
// Flash attention (online softmax), one wave = 16 queries of one (b, head).
// Computes S^T = K @ Q^T so softmax reduction is in-lane + one shfl_xor(16),
// and exp(S^T) C-fragments map directly onto the P^T B-fragment of the PV
// WMMA (O^T = V^T @ P^T) with 16 shfl_xor(16)s — no LDS transpose.
//   NF = DH/32 (K-dim fragments).  Each wave produces 8 feature tiles
//   (128 features); MHA2 (DH=256) uses FS=2 waves per (b,h,qtile).
// ---------------------------------------------------------------------------
template <int NF>
__global__ void attn_kernel(const __bf16* __restrict__ Q, const __bf16* __restrict__ Kb,
                            const __bf16* __restrict__ Vt, const int* __restrict__ ids,
                            __bf16* __restrict__ O,
                            int S, int H, int Dt, float scale, int FS) {
    constexpr int DH = NF * 32;
    const int lane = threadIdx.x & 31;
    const int wslot = threadIdx.x >> 5;
    int w = blockIdx.x * (blockDim.x >> 5) + wslot;

    const int qtiles = S / 16;
    const int fh = w % FS;  w /= FS;
    const int qt = w % qtiles;  w /= qtiles;
    const int h = w % H;  w /= H;
    const int b = w;

    const int n = lane & 15;
    const int half8 = (lane >= 16) ? 8 : 0;
    const int half16 = (lane >= 16) ? 16 : 0;
    const bool lo16 = lane < 16;

    __shared__ float smask[8][512];
    for (int s = lane; s < S; s += 32)
        smask[wslot][s] = (ids[b * S + s] != 0) ? 0.0f : -1e30f;
    __syncthreads();

    // Q fragments (B-layout, loop-invariant): lane n = query
    const int qrow = b * S + qt * 16;
    bhalf16 qf[NF];
    const __bf16* qbase = Q + (size_t)(qrow + n) * Dt + h * DH + half16;
#pragma unroll
    for (int f = 0; f < NF; ++f) qf[f] = ldB16(qbase + f * 32);

    floatx8 o[8];
#pragma unroll
    for (int j = 0; j < 8; ++j) o[j] = (floatx8){};
    float rowmax = -1e30f;
    float rowsum = 0.0f;

    const __bf16* vbase =
        Vt + ((size_t)b * Dt + h * DH + fh * 128 + n) * S + half16;

    for (int kb = 0; kb < S; kb += 32) {
        // scores S^T for 32 keys x 16 queries (two 16x16 tiles)
        floatx8 st0 = {}, st1 = {};
        const __bf16* krow0 = Kb + (size_t)(b * S + kb + n) * Dt + h * DH + half16;
        const __bf16* krow1 = krow0 + (size_t)16 * Dt;
#pragma unroll
        for (int f = 0; f < NF; ++f) {
            bhalf16 a0 = ldA16(krow0 + f * 32);
            bhalf16 a1 = ldA16(krow1 + f * 32);
            st0 = WMMA_BF16(a0, qf[f], st0);
            st1 = WMMA_BF16(a1, qf[f], st1);
        }
        // scale + key-padding mask (rows of S^T are keys)
        float tmax = -1e30f;
#pragma unroll
        for (int r = 0; r < 8; ++r) {
            st0[r] = st0[r] * scale + smask[wslot][kb + r + half8];
            st1[r] = st1[r] * scale + smask[wslot][kb + 16 + r + half8];
            tmax = fmaxf(tmax, fmaxf(st0[r], st1[r]));
        }
        tmax = fmaxf(tmax, __shfl_xor(tmax, 16, 32));
        const float nm = fmaxf(rowmax, tmax);
        const float corr = __expf(rowmax - nm);
        rowmax = nm;

        float p0[8], p1[8];
        float tsum = 0.0f;
#pragma unroll
        for (int r = 0; r < 8; ++r) {
            p0[r] = __expf(st0[r] - nm);
            p1[r] = __expf(st1[r] - nm);
            tsum += p0[r] + p1[r];
        }
        tsum += __shfl_xor(tsum, 16, 32);
        rowsum = rowsum * corr + tsum;

#pragma unroll
        for (int j = 0; j < 8; ++j)
#pragma unroll
            for (int r = 0; r < 8; ++r) o[j][r] *= corr;

        // build P^T B-fragment: lane n = query, elements = 32 keys
        bhalf16 bp;
#pragma unroll
        for (int e = 0; e < 8; ++e) {
            float t0x = __shfl_xor(p0[e], 16, 32);
            float t1x = __shfl_xor(p1[e], 16, 32);
            bp[e]     = (__bf16)(lo16 ? p0[e] : t1x);
            bp[e + 8] = (__bf16)(lo16 ? t0x : p1[e]);
        }
        // O^T += V^T @ P^T   (A: lane m = feature, contiguous keys from Vt)
#pragma unroll
        for (int j = 0; j < 8; ++j) {
            bhalf16 av = ldA16(vbase + (size_t)j * 16 * S + kb);
            o[j] = WMMA_BF16(av, bp, o[j]);
        }
    }

    const float inv = (rowsum > 0.0f) ? (1.0f / rowsum) : 0.0f;
    // O^T C layout: lane n = query (fixed row), 8 consecutive features per lane
    __bf16* orow = O + (size_t)(qrow + n) * Dt + h * DH + fh * 128 + half8;
#pragma unroll
    for (int j = 0; j < 8; ++j) {
        bhalf8 st;
#pragma unroll
        for (int r = 0; r < 8; ++r) st[r] = (__bf16)(o[j][r] * inv);
        *(bhalf8*)(orow + j * 16) = st;
    }
}

// ---------------------------------------------------------------------------
// Final: h3 = h2 + h0; masked mean-pool over tokens; LayerNorm; dot lin_w.
// One block (256 threads) per batch element; D must be 1024.
// ---------------------------------------------------------------------------
__global__ void finalize_kernel(const float* __restrict__ h2, const float* __restrict__ h0,
                                const int* __restrict__ ids, const float* __restrict__ gw,
                                const float* __restrict__ gb, const float* __restrict__ lw,
                                const float* __restrict__ lb, float* __restrict__ out,
                                int S, int D) {
    const int b = blockIdx.x, tid = threadIdx.x, NT = blockDim.x;
    __shared__ float red[256];
    __shared__ float pooled[1024];

    float cnt = 0.0f;
    for (int s = tid; s < S; s += NT) cnt += (ids[b * S + s] != 0) ? 1.0f : 0.0f;
    red[tid] = cnt; __syncthreads();
    for (int st = NT >> 1; st > 0; st >>= 1) {
        if (tid < st) red[tid] += red[tid + st];
        __syncthreads();
    }
    const float count = fmaxf(red[0], 1.0f);
    __syncthreads();

    float acc[4] = {0.f, 0.f, 0.f, 0.f};
    for (int s = 0; s < S; ++s) {
        if (ids[b * S + s] != 0) {
            const size_t base = ((size_t)b * S + s) * (size_t)D;
#pragma unroll
            for (int j = 0; j < 4; ++j)
                acc[j] += h2[base + tid + j * 256] + h0[base + tid + j * 256];
        }
    }
    float lsum = 0.0f;
#pragma unroll
    for (int j = 0; j < 4; ++j) {
        float p = acc[j] / count;
        pooled[tid + j * 256] = p;
        lsum += p;
    }
    red[tid] = lsum; __syncthreads();
    for (int st = NT >> 1; st > 0; st >>= 1) {
        if (tid < st) red[tid] += red[tid + st];
        __syncthreads();
    }
    const float mu = red[0] / (float)D;
    __syncthreads();

    float vsum = 0.0f;
#pragma unroll
    for (int j = 0; j < 4; ++j) {
        float d = pooled[tid + j * 256] - mu;
        vsum += d * d;
    }
    red[tid] = vsum; __syncthreads();
    for (int st = NT >> 1; st > 0; st >>= 1) {
        if (tid < st) red[tid] += red[tid + st];
        __syncthreads();
    }
    const float rstd = rsqrtf(red[0] / (float)D + 1e-5f);
    __syncthreads();

    float dsum = 0.0f;
#pragma unroll
    for (int j = 0; j < 4; ++j) {
        int d = tid + j * 256;
        float nv = (pooled[d] - mu) * rstd * gw[d] + gb[d];
        dsum += nv * lw[d];
    }
    red[tid] = dsum; __syncthreads();
    for (int st = NT >> 1; st > 0; st >>= 1) {
        if (tid < st) red[tid] += red[tid + st];
        __syncthreads();
    }
    if (tid == 0) out[b] = red[0] + lb[0];
}

// ---------------------------------------------------------------------------
// Host-side orchestration
// ---------------------------------------------------------------------------
extern "C" void kernel_launch(void* const* d_in, const int* in_sizes, int n_in,
                              void* d_out, int out_size, void* d_ws, size_t ws_size,
                              hipStream_t stream) {
    (void)in_sizes; (void)n_in; (void)out_size; (void)ws_size;
    const int B = 64, S = 512, D = 1024;
    const int M = B * S;

    const int*   x   = (const int*)d_in[0];
    const float* emb = (const float*)d_in[1];
    const float* pos = (const float*)d_in[2];
    const float* W[8] = {
        (const float*)d_in[3],  (const float*)d_in[5],  (const float*)d_in[7],  (const float*)d_in[9],
        (const float*)d_in[11], (const float*)d_in[13], (const float*)d_in[15], (const float*)d_in[17]
    };
    const float* Bias[8] = {
        (const float*)d_in[4],  (const float*)d_in[6],  (const float*)d_in[8],  (const float*)d_in[10],
        (const float*)d_in[12], (const float*)d_in[14], (const float*)d_in[16], (const float*)d_in[18]
    };
    const float* ng = (const float*)d_in[19];
    const float* nb = (const float*)d_in[20];
    const float* lw = (const float*)d_in[21];
    const float* lb = (const float*)d_in[22];
    float* out = (float*)d_out;

    // Workspace carve-out
    char* ws = (char*)d_ws;
    size_t off = 0;
    auto carve = [&](size_t bytes) -> char* {
        char* p = ws + off;
        off += (bytes + 255) & ~(size_t)255;
        return p;
    };
    float*  h0f = (float*)carve((size_t)M * D * sizeof(float));   // embedding (residual)
    float*  hcf = (float*)carve((size_t)M * D * sizeof(float));   // current f32 activations
    __bf16* hb  = (__bf16*)carve((size_t)M * D * sizeof(__bf16)); // current bf16 activations
    __bf16* Qb  = (__bf16*)carve((size_t)M * D * sizeof(__bf16));
    __bf16* Kb  = (__bf16*)carve((size_t)M * D * sizeof(__bf16));
    __bf16* Vt  = (__bf16*)carve((size_t)M * D * sizeof(__bf16)); // V transposed per batch
    __bf16* Ab  = (__bf16*)carve((size_t)M * D * sizeof(__bf16)); // attention output
    __bf16* wt[8];
    for (int i = 0; i < 8; ++i) wt[i] = (__bf16*)carve((size_t)D * D * sizeof(__bf16));

    // Weight transpose + bf16
    {
        const int tot = D * D, blk = 256, grd = (tot + blk - 1) / blk;
        for (int i = 0; i < 8; ++i)
            wtr_kernel<<<grd, blk, 0, stream>>>(W[i], wt[i], D, D);
    }

    // Embedding
    embed_kernel<<<M, 256, 0, stream>>>(x, emb, pos, h0f, hb, S, D);

    const int gblocks = (M / 256) * (D / 64);   // 256x64 block tiles, 8 waves each

    // ---- MHA block 1 (8 heads, d=128) ----
    gemm_bf16<<<gblocks, 256, 0, stream>>>(hb, wt[0], Bias[0], nullptr, nullptr, Qb, nullptr, M, D, D, S);
    gemm_bf16<<<gblocks, 256, 0, stream>>>(hb, wt[1], Bias[1], nullptr, nullptr, Kb, nullptr, M, D, D, S);
    gemm_bf16<<<gblocks, 256, 0, stream>>>(hb, wt[2], Bias[2], nullptr, nullptr, nullptr, Vt, M, D, D, S);
    attn_kernel<4><<<(B * 8 * (S / 16) * 1) / 8, 256, 0, stream>>>(
        Qb, Kb, Vt, x, Ab, S, 8, D, 0.088388347648318447f /* 1/sqrt(128) */, 1);
    gemm_bf16<<<gblocks, 256, 0, stream>>>(Ab, wt[3], Bias[3], h0f, hcf, hb, nullptr, M, D, D, S);

    // ---- MHA block 2 (4 heads, d=256) ----
    gemm_bf16<<<gblocks, 256, 0, stream>>>(hb, wt[4], Bias[4], nullptr, nullptr, Qb, nullptr, M, D, D, S);
    gemm_bf16<<<gblocks, 256, 0, stream>>>(hb, wt[5], Bias[5], nullptr, nullptr, Kb, nullptr, M, D, D, S);
    gemm_bf16<<<gblocks, 256, 0, stream>>>(hb, wt[6], Bias[6], nullptr, nullptr, nullptr, Vt, M, D, D, S);
    attn_kernel<8><<<(B * 4 * (S / 16) * 2) / 8, 256, 0, stream>>>(
        Qb, Kb, Vt, x, Ab, S, 4, D, 0.0625f /* 1/sqrt(256) */, 2);
    gemm_bf16<<<gblocks, 256, 0, stream>>>(Ab, wt[7], Bias[7], hcf, hcf, nullptr, nullptr, M, D, D, S);

    // ---- pooling + LayerNorm + linear ----
    finalize_kernel<<<B, 256, 0, stream>>>(hcf, h0f, x, ng, nb, lw, lb, out, S, D);
}
